// GINECombined_v2_13262859010609
// MI455X (gfx1250) — matrix operand order repaired and checked by reading, not compiled, not run on hardware
//
#include <hip/hip_runtime.h>
#include <hip/hip_bf16.h>
#include <math.h>

// ---------------------------------------------------------------------------
// GINE GNN forward for MI455X (gfx1250, wave32).
//  - Edge aggregation: bandwidth/atomic bound -> scalar f32 + LDS-staged We.
//  - Node MLP / gate GEMMs: v_wmma_f32_16x16x32_bf16; weights staged into LDS
//    via the Tensor Data Mover (tensor_load_to_lds + s_wait_tensorcnt);
//    double-buffered LDS fragments to overlap ds_load latency with WMMA.
// ---------------------------------------------------------------------------

#define N_NODES 100000
#define N_EDGES 1600000
#define N_GRAPH 128

typedef __bf16 bf16_t;
typedef __attribute__((ext_vector_type(16))) __bf16 v16bf;
typedef __attribute__((ext_vector_type(8)))  __bf16 v8bf;
typedef __attribute__((ext_vector_type(4)))  __bf16 v4bf;
typedef __attribute__((ext_vector_type(8)))  float  v8f;
typedef unsigned int u32;
typedef __attribute__((ext_vector_type(4))) u32 v4u;
typedef __attribute__((ext_vector_type(8))) int v8i;
typedef __attribute__((ext_vector_type(4))) int v4i;

#define LDA 136           // bf16 elements per LDS tile row (padded)
#define LDF 132           // f32  elements per LDS tile row (padded)
#define TILE_BYTES (16 * LDF * 4)   // 8448 B: f32 tile dominates the union

__device__ __forceinline__ float gelu_f(float x) {
  return 0.5f * x * (1.0f + erff(x * 0.70710678118654752f));
}

__device__ __forceinline__ void atomicMaxFloat(float* addr, float val) {
  if (val >= 0.0f) atomicMax((int*)addr, __float_as_int(val));
  else             atomicMin((unsigned int*)addr, __float_as_uint(val));
}

// ---------------------------------------------------------------------------
// TDM: DMA a 128x128 bf16 tile (row-major, stride 128) from global into LDS.
// Descriptor per CDNA5 ISA ch.8 (D# group0/group1). Issued per-wave;
// completion tracked with TENSORcnt.
// ---------------------------------------------------------------------------
#if __has_builtin(__builtin_amdgcn_tensor_load_to_lds)
#define HAVE_TDM 1
__device__ __forceinline__ void tdm_load_128x128_bf16(const bf16_t* gsrc, bf16_t* lds_dst) {
  // LDS flat address: low 32 bits are the LDS byte offset (ISA 10.2 aperture rule)
  const u32 lds_addr = (u32)(unsigned long long)(uintptr_t)lds_dst;
  const unsigned long long ga = (unsigned long long)(uintptr_t)gsrc;

  v4u g0;
  g0[0] = 1u;                                             // count=1 (valid), load, user
  g0[1] = lds_addr;                                       // lds_addr[31:0]
  g0[2] = (u32)(ga & 0xFFFFFFFFu);                        // global_addr[31:0]
  g0[3] = (u32)((ga >> 32) & 0x01FFFFFFu) | (2u << 30);   // global_addr[56:32], type=2

  const u32 dim0 = 128u, dim1 = 128u, t0 = 128u, t1 = 128u, stride0 = 128u;
  v8i g1;
  g1[0] = (int)(1u << 16);                                // wg_mask=0, data_size=1 (2B)
  g1[1] = (int)((dim0 & 0xFFFFu) << 16);                  // tensor_dim0[15:0]
  g1[2] = (int)(((dim0 >> 16) & 0xFFFFu) | ((dim1 & 0xFFFFu) << 16));
  g1[3] = (int)(((dim1 >> 16) & 0xFFFFu) | ((t0 & 0xFFFFu) << 16)); // tile_dim0
  g1[4] = (int)(t1 & 0xFFFFu);                            // tile_dim1, tile_dim2=0
  g1[5] = (int)stride0;                                   // tensor_dim0_stride[31:0]
  g1[6] = 0;                                              // stride0[47:32], stride1[15:0]
  g1[7] = 0;                                              // stride1[47:16]

  v4i gz = {0, 0, 0, 0};
#if defined(__clang_major__) && __clang_major__ >= 23
  v8i gz8 = {0, 0, 0, 0, 0, 0, 0, 0};
  __builtin_amdgcn_tensor_load_to_lds(g0, g1, gz, gz, gz8, 0);
#else
  __builtin_amdgcn_tensor_load_to_lds(g0, g1, gz, gz, 0);
#endif
}
__device__ __forceinline__ void tdm_wait() {
#if __has_builtin(__builtin_amdgcn_s_wait_tensorcnt)
  __builtin_amdgcn_s_wait_tensorcnt((short)0);
#endif
}
#else
#define HAVE_TDM 0
#endif

// Cooperative fallback copy (also used if TDM builtins are unavailable)
__device__ __forceinline__ void coop_copy_32kb(const bf16_t* gsrc, bf16_t* lds_dst,
                                               int tid, int nthreads) {
  const u32* s = (const u32*)gsrc;
  u32* d = (u32*)lds_dst;
  for (int i = tid; i < 128 * 128 / 2; i += nthreads) d[i] = s[i];
}

// ---------------------------------------------------------------------------
// Utility kernels
// ---------------------------------------------------------------------------
__global__ void k_zero(float* __restrict__ p, long long n) {
  long long i = (long long)blockIdx.x * blockDim.x + threadIdx.x;
  long long stride = (long long)gridDim.x * blockDim.x;
  for (; i < n; i += stride) p[i] = 0.0f;
}

__global__ void k_init_pool(float* __restrict__ gmax, float* __restrict__ denom) {
  int t = threadIdx.x;
  if (t < N_GRAPH) { gmax[t] = -__builtin_inff(); denom[t] = 0.0f; }
}

// f32 [128 x 128] (k-major) -> bf16 transposed [n][k]
__global__ void k_conv_t(const float* __restrict__ W, bf16_t* __restrict__ WT) {
  int idx = blockIdx.x * blockDim.x + threadIdx.x;
  if (idx >= 128 * 128) return;
  int n = idx & 127, k = idx >> 7;
  WT[n * 128 + k] = (bf16_t)W[k * 128 + n];
}

// ---------------------------------------------------------------------------
// Edge kernel: msg = relu(x[src] + edge_attr @ We + be); atomic-add into agg[dst]
// ---------------------------------------------------------------------------
__global__ __launch_bounds__(256) void k_edge(
    const float* __restrict__ x, const int* __restrict__ ei,
    const float* __restrict__ ea, const float* __restrict__ We,
    const float* __restrict__ be, float* __restrict__ agg)
{
  __shared__ float sWe[16 * 128];
  __shared__ float sBe[128];
  for (int i = threadIdx.x; i < 16 * 128; i += 256) sWe[i] = We[i];
  if (threadIdx.x < 128) sBe[threadIdx.x] = be[threadIdx.x];
  __syncthreads();

  const int e = blockIdx.x * 8 + (threadIdx.x >> 5);
  if (e >= N_EDGES) return;
  const int lane = threadIdx.x & 31;
  const int f = lane * 4;
  const int s = ei[e];
  const int d = ei[N_EDGES + e];

  float a0 = sBe[f + 0], a1 = sBe[f + 1], a2 = sBe[f + 2], a3 = sBe[f + 3];
  const float* eap = ea + (size_t)e * 16;
#pragma unroll
  for (int k = 0; k < 16; ++k) {
    const float ev = eap[k];
    const float4 wv = *(const float4*)(sWe + k * 128 + f);
    a0 = fmaf(ev, wv.x, a0); a1 = fmaf(ev, wv.y, a1);
    a2 = fmaf(ev, wv.z, a2); a3 = fmaf(ev, wv.w, a3);
  }
  const float4 xv = *(const float4*)(x + (size_t)s * 128 + f);
  a0 = fmaxf(xv.x + a0, 0.0f); a1 = fmaxf(xv.y + a1, 0.0f);
  a2 = fmaxf(xv.z + a2, 0.0f); a3 = fmaxf(xv.w + a3, 0.0f);

  float* ap = agg + (size_t)d * 128 + f;
  atomicAdd(ap + 0, a0); atomicAdd(ap + 1, a1);
  atomicAdd(ap + 2, a2); atomicAdd(ap + 3, a3);
}

// ---------------------------------------------------------------------------
// WMMA fragment helpers (ISA 7.12.2, 16-bit layouts, wave32)
// A 16x32 (MxK): lane<16: K=0..7,16..23 ; lane>=16: K=8..15,24..31 ; M=lane&15
// B 32x16 (KxN): lane<16: K=0..7,16..23 ; lane>=16: K=8..15,24..31 ; N=lane&15
// C/D 16x16 f32: VGPR r -> M = r + (lane>=16 ? 8:0), N = lane&15
// ---------------------------------------------------------------------------
__device__ __forceinline__ v16bf load_a_frag(const bf16_t* At, int m, int hi, int kt) {
  const bf16_t* ap = At + m * LDA + kt * 32 + hi * 8;
  const v8bf a0 = *(const v8bf*)(ap);
  const v8bf a1 = *(const v8bf*)(ap + 16);
  v16bf a;
#pragma unroll
  for (int i = 0; i < 8; ++i) { a[i] = a0[i]; a[8 + i] = a1[i]; }
  return a;
}

__device__ __forceinline__ v16bf load_b_frag(const bf16_t* WT, int n, int hi, int kt) {
  const bf16_t* bp = WT + (size_t)n * 128 + kt * 32 + hi * 8;
  const v8bf b0 = *(const v8bf*)(bp);
  const v8bf b1 = *(const v8bf*)(bp + 16);
  v16bf b;
#pragma unroll
  for (int i = 0; i < 8; ++i) { b[i] = b0[i]; b[8 + i] = b1[i]; }
  return b;
}

// 16x128 @ 128x128 GEMM accumulate, double-buffered LDS fragments.
// acc[nt] must be pre-initialized (bias). A from At (LDS tile), B from sW (LDS).
__device__ __forceinline__ void gemm_tile_128(v8f acc[8], const bf16_t* At,
                                              const bf16_t* sW, int m, int hi) {
  v16bf a_cur = load_a_frag(At, m, hi, 0);
  v16bf b_cur = load_b_frag(sW, m, hi, 0);        // kt=0, nt=0
#pragma unroll
  for (int step = 0; step < 32; ++step) {
    const int nt = step & 7;
    const v16bf a_use = a_cur;
    const v16bf b_use = b_cur;
    const int ns = step + 1;
    if (ns < 32) {
      const int nkt = ns >> 3, nnt = ns & 7;
      if (nnt == 0) a_cur = load_a_frag(At, m, hi, nkt);
      b_cur = load_b_frag(sW, nnt * 16 + m, hi, nkt);
    }
    acc[nt] = __builtin_amdgcn_wmma_f32_16x16x32_bf16(
        false, a_use, false, b_use, (short)0, acc[nt], false, false);
  }
}

// ---------------------------------------------------------------------------
// Fused GINE node MLP: h=(1+eps)*x+agg; out = gelu(LN(gelu(h@W1+b1)@W2+b2))
// 320 threads = 10 waves, each owning one 16x128 tile (grid 625 => 6250 tiles
// == N/16 exactly; no divergent barriers). W1/W2 staged in LDS via TDM.
// ---------------------------------------------------------------------------
__global__ __launch_bounds__(320) void k_gine_mlp(
    const float* __restrict__ xin, const float* __restrict__ agg,
    const float* __restrict__ eps_p,
    const bf16_t* __restrict__ W1T, const float* __restrict__ b1,
    const bf16_t* __restrict__ W2T, const float* __restrict__ b2,
    const float* __restrict__ ln_g, const float* __restrict__ ln_b,
    float* __restrict__ out)
{
  __shared__ __align__(16) char   sTile[10][TILE_BYTES];  // bf16 A-tile / f32 LN tile
  __shared__ __align__(16) bf16_t sW1[128 * 128];
  __shared__ __align__(16) bf16_t sW2[128 * 128];

  const int w    = threadIdx.x >> 5;
  const int lane = threadIdx.x & 31;
  const int row0 = (blockIdx.x * 10 + w) * 16;
  const int m  = lane & 15;
  const int hi = lane >> 4;
  const float epsv = 1.0f + eps_p[0];
  bf16_t* At = (bf16_t*)sTile[w];
  float*  Ft = (float*)sTile[w];

#if HAVE_TDM
  if (w == 0) {
    tdm_load_128x128_bf16(W1T, sW1);
    tdm_load_128x128_bf16(W2T, sW2);
    tdm_wait();
  }
#else
  coop_copy_32kb(W1T, sW1, threadIdx.x, 320);
  coop_copy_32kb(W2T, sW2, threadIdx.x, 320);
#endif

  // Stage h = (1+eps)*x + agg as bf16 A tile (each lane: 4 cols x 16 rows)
  {
    const int c = lane * 4;
#pragma unroll
    for (int r = 0; r < 16; ++r) {
      const float4 xv = *(const float4*)(xin + (size_t)(row0 + r) * 128 + c);
      const float4 av = *(const float4*)(agg + (size_t)(row0 + r) * 128 + c);
      v4bf hv;
      hv[0] = (bf16_t)(epsv * xv.x + av.x);
      hv[1] = (bf16_t)(epsv * xv.y + av.y);
      hv[2] = (bf16_t)(epsv * xv.z + av.z);
      hv[3] = (bf16_t)(epsv * xv.w + av.w);
      *(v4bf*)(At + r * LDA + c) = hv;
    }
  }
  __syncthreads();   // weights staged + all tiles in place

  // ---- GEMM1: acc = h @ W1 + b1 ----
  v8f acc[8];
#pragma unroll
  for (int nt = 0; nt < 8; ++nt) {
    const float bv = b1[nt * 16 + m];
#pragma unroll
    for (int i = 0; i < 8; ++i) acc[nt][i] = bv;
  }
  gemm_tile_128(acc, At, sW1, m, hi);

  // gelu -> bf16 tile (A for GEMM2); per-wave LDS ops are in-order
#pragma unroll
  for (int nt = 0; nt < 8; ++nt) {
#pragma unroll
    for (int r = 0; r < 8; ++r) {
      At[(r + hi * 8) * LDA + nt * 16 + m] = (bf16_t)gelu_f(acc[nt][r]);
    }
  }

  // ---- GEMM2: acc = gelu(...) @ W2 + b2 ----
#pragma unroll
  for (int nt = 0; nt < 8; ++nt) {
    const float bv = b2[nt * 16 + m];
#pragma unroll
    for (int i = 0; i < 8; ++i) acc[nt][i] = bv;
  }
  gemm_tile_128(acc, At, sW2, m, hi);

  // spill f32 tile for LayerNorm (overlays the bf16 tile; in-order per wave)
#pragma unroll
  for (int nt = 0; nt < 8; ++nt) {
#pragma unroll
    for (int r = 0; r < 8; ++r) {
      Ft[(r + hi * 8) * LDF + nt * 16 + m] = acc[nt][r];
    }
  }

  // LayerNorm + GELU: lane -> row (lane&15), half (lane>>4) of 64 cols
  const int rr = m;
  const float* rp = Ft + rr * LDF + hi * 64;
  float s = 0.0f, s2 = 0.0f;
#pragma unroll
  for (int j = 0; j < 16; ++j) {
    const float4 v = *(const float4*)(rp + j * 4);
    s  += v.x + v.y + v.z + v.w;
    s2 += v.x * v.x + v.y * v.y + v.z * v.z + v.w * v.w;
  }
  s  += __shfl_xor(s, 16, 32);
  s2 += __shfl_xor(s2, 16, 32);
  const float mu   = s * (1.0f / 128.0f);
  const float rstd = rsqrtf(s2 * (1.0f / 128.0f) - mu * mu + 1e-5f);

  float* op = out + (size_t)(row0 + rr) * 128 + hi * 64;
#pragma unroll
  for (int j = 0; j < 16; ++j) {
    const int c = hi * 64 + j * 4;
    const float4 v = *(const float4*)(rp + j * 4);
    float4 o;
    o.x = gelu_f((v.x - mu) * rstd * ln_g[c + 0] + ln_b[c + 0]);
    o.y = gelu_f((v.y - mu) * rstd * ln_g[c + 1] + ln_b[c + 1]);
    o.z = gelu_f((v.z - mu) * rstd * ln_g[c + 2] + ln_b[c + 2]);
    o.w = gelu_f((v.w - mu) * rstd * ln_g[c + 3] + ln_b[c + 3]);
    *(float4*)(op + j * 4) = o;
  }
}

// ---------------------------------------------------------------------------
// Gate kernel: gate[n] = gelu(h@Wg1+bg1)@Wg2 + bg2, plus segment atomic max.
// ---------------------------------------------------------------------------
__global__ __launch_bounds__(320) void k_gate(
    const float* __restrict__ h, const bf16_t* __restrict__ Wg1T,
    const float* __restrict__ bg1, const float* __restrict__ wg2,
    const float* __restrict__ bg2, const int* __restrict__ batch,
    float* __restrict__ gate, float* __restrict__ gmax)
{
  __shared__ __align__(16) bf16_t sA[10][16 * LDA];
  __shared__ __align__(16) bf16_t sW[128 * 128];

  const int w    = threadIdx.x >> 5;
  const int lane = threadIdx.x & 31;
  const int row0 = (blockIdx.x * 10 + w) * 16;
  const int m  = lane & 15;
  const int hi = lane >> 4;
  bf16_t* At = sA[w];

#if HAVE_TDM
  if (w == 0) {
    tdm_load_128x128_bf16(Wg1T, sW);
    tdm_wait();
  }
#else
  coop_copy_32kb(Wg1T, sW, threadIdx.x, 320);
#endif

  {
    const int c = lane * 4;
#pragma unroll
    for (int r = 0; r < 16; ++r) {
      const float4 xv = *(const float4*)(h + (size_t)(row0 + r) * 128 + c);
      v4bf hv;
      hv[0] = (bf16_t)xv.x; hv[1] = (bf16_t)xv.y;
      hv[2] = (bf16_t)xv.z; hv[3] = (bf16_t)xv.w;
      *(v4bf*)(At + r * LDA + c) = hv;
    }
  }
  __syncthreads();

  v8f acc[8];
#pragma unroll
  for (int nt = 0; nt < 8; ++nt) {
    const float bv = bg1[nt * 16 + m];
#pragma unroll
    for (int i = 0; i < 8; ++i) acc[nt][i] = bv;
  }
  gemm_tile_128(acc, At, sW, m, hi);

  // gate = gelu(acc) . wg2, reduced across the 16 lanes of each half-wave
  float p[8];
#pragma unroll
  for (int r = 0; r < 8; ++r) p[r] = 0.0f;
#pragma unroll
  for (int nt = 0; nt < 8; ++nt) {
    const float wv = wg2[nt * 16 + m];
#pragma unroll
    for (int r = 0; r < 8; ++r) p[r] += gelu_f(acc[nt][r]) * wv;
  }
#pragma unroll
  for (int off = 1; off < 16; off <<= 1) {
#pragma unroll
    for (int r = 0; r < 8; ++r) p[r] += __shfl_xor(p[r], off, 32);
  }
  if (m == 0) {
    const float bg = bg2[0];
#pragma unroll
    for (int r = 0; r < 8; ++r) {
      const int node = row0 + r + hi * 8;
      const float gv = p[r] + bg;
      gate[node] = gv;
      atomicMaxFloat(&gmax[batch[node]], gv);
    }
  }
}

// ---------------------------------------------------------------------------
// Softmax pieces
// ---------------------------------------------------------------------------
__global__ void k_softmax1(const float* __restrict__ gate, const float* __restrict__ gmax,
                           const int* __restrict__ batch,
                           float* __restrict__ ex, float* __restrict__ denom) {
  const int n = blockIdx.x * blockDim.x + threadIdx.x;
  if (n >= N_NODES) return;
  const int g = batch[n];
  const float e = expf(gate[n] - gmax[g]);
  ex[n] = e;
  atomicAdd(&denom[g], e);
}

__global__ void k_pool(const float* __restrict__ h, const float* __restrict__ ex,
                       const float* __restrict__ denom, const int* __restrict__ batch,
                       float* __restrict__ pooled) {
  const long long tid = (long long)blockIdx.x * blockDim.x + threadIdx.x;
  const int n = (int)(tid >> 5);
  if (n >= N_NODES) return;
  const int c = ((int)tid & 31) * 4;
  const int g = batch[n];
  const float wv = ex[n] / denom[g];
  const float4 v = *(const float4*)(h + (size_t)n * 128 + c);
  float* p = pooled + g * 128 + c;
  atomicAdd(p + 0, wv * v.x); atomicAdd(p + 1, wv * v.y);
  atomicAdd(p + 2, wv * v.z); atomicAdd(p + 3, wv * v.w);
}

// ---------------------------------------------------------------------------
// Classifier: one block per graph (G=128), tiny MLP.
// ---------------------------------------------------------------------------
__global__ __launch_bounds__(256) void k_classifier(
    const float* __restrict__ pooled, const float* __restrict__ gf,
    const float* __restrict__ cg, const float* __restrict__ cb,
    const float* __restrict__ Wc1, const float* __restrict__ bc1,
    const float* __restrict__ Wc2, const float* __restrict__ bc2,
    const float* __restrict__ Wc3, const float* __restrict__ bc3,
    float* __restrict__ out)
{
  __shared__ float z[160];
  __shared__ float z1[512];
  __shared__ float z2[128];
  __shared__ float red[2];
  const int g = blockIdx.x;
  const int t = threadIdx.x;

  if (t < 128)      z[t] = pooled[g * 128 + t];
  else if (t < 160) z[t] = gf[g * 32 + (t - 128)];
  __syncthreads();

  if (t == 0) {
    float s = 0.0f, s2 = 0.0f;
    for (int i = 0; i < 160; ++i) { s += z[i]; s2 += z[i] * z[i]; }
    const float mu = s / 160.0f;
    red[0] = mu;
    red[1] = rsqrtf(s2 / 160.0f - mu * mu + 1e-5f);
  }
  __syncthreads();
  if (t < 160) z[t] = (z[t] - red[0]) * red[1] * cg[t] + cb[t];
  __syncthreads();

  for (int o = t; o < 512; o += 256) {
    float a = bc1[o];
    for (int k = 0; k < 160; ++k) a = fmaf(z[k], Wc1[k * 512 + o], a);
    z1[o] = gelu_f(a);
  }
  __syncthreads();

  if (t < 128) {
    float a = bc2[t];
    for (int k = 0; k < 512; ++k) a = fmaf(z1[k], Wc2[k * 128 + t], a);
    z2[t] = gelu_f(a);
  }
  __syncthreads();

  if (t < 2) {
    float a = bc3[t];
    for (int k = 0; k < 128; ++k) a = fmaf(z2[k], Wc3[k * 2 + t], a);
    out[g * 2 + t] = a;
  }
}

// ---------------------------------------------------------------------------
// Launcher
// ---------------------------------------------------------------------------
extern "C" void kernel_launch(void* const* d_in, const int* in_sizes, int n_in,
                              void* d_out, int out_size, void* d_ws, size_t ws_size,
                              hipStream_t stream) {
  (void)in_sizes; (void)n_in; (void)out_size; (void)ws_size;

  const float* x     = (const float*)d_in[0];
  const int*   ei    = (const int*)d_in[1];
  const float* ea    = (const float*)d_in[2];
  const int*   batch = (const int*)d_in[3];
  const float* gf    = (const float*)d_in[4];
  const float* eps0  = (const float*)d_in[5];
  const float* We0   = (const float*)d_in[6];
  const float* be0   = (const float*)d_in[7];
  const float* W1_0  = (const float*)d_in[8];
  const float* b1_0  = (const float*)d_in[9];
  const float* W2_0  = (const float*)d_in[10];
  const float* b2_0  = (const float*)d_in[11];
  const float* lg0   = (const float*)d_in[12];
  const float* lb0   = (const float*)d_in[13];
  const float* eps1  = (const float*)d_in[14];
  const float* We1   = (const float*)d_in[15];
  const float* be1   = (const float*)d_in[16];
  const float* W1_1  = (const float*)d_in[17];
  const float* b1_1  = (const float*)d_in[18];
  const float* W2_1  = (const float*)d_in[19];
  const float* b2_1  = (const float*)d_in[20];
  const float* lg1   = (const float*)d_in[21];
  const float* lb1   = (const float*)d_in[22];
  const float* Wg1   = (const float*)d_in[23];
  const float* bg1   = (const float*)d_in[24];
  const float* Wg2   = (const float*)d_in[25];
  const float* bg2   = (const float*)d_in[26];
  const float* cg    = (const float*)d_in[27];
  const float* cb    = (const float*)d_in[28];
  const float* Wc1   = (const float*)d_in[29];
  const float* bc1   = (const float*)d_in[30];
  const float* Wc2   = (const float*)d_in[31];
  const float* bc2   = (const float*)d_in[32];
  const float* Wc3   = (const float*)d_in[33];
  const float* bc3   = (const float*)d_in[34];

  char* ws = (char*)d_ws;
  size_t off = 0;
  auto alloc = [&](size_t bytes) -> char* {
    char* p = ws + off;
    off = (off + bytes + 255) & ~(size_t)255;
    return p;
  };
  float*  agg0   = (float*)alloc((size_t)N_NODES * 128 * sizeof(float));
  float*  agg1   = (float*)alloc((size_t)N_NODES * 128 * sizeof(float));
  float*  gate   = (float*)alloc((size_t)N_NODES * sizeof(float));
  float*  exb    = (float*)alloc((size_t)N_NODES * sizeof(float));
  float*  gmax   = (float*)alloc(N_GRAPH * sizeof(float));
  float*  denom  = (float*)alloc(N_GRAPH * sizeof(float));
  float*  pooled = (float*)alloc((size_t)N_GRAPH * 128 * sizeof(float));
  bf16_t* W1_0T  = (bf16_t*)alloc(128 * 128 * sizeof(bf16_t));
  bf16_t* W2_0T  = (bf16_t*)alloc(128 * 128 * sizeof(bf16_t));
  bf16_t* W1_1T  = (bf16_t*)alloc(128 * 128 * sizeof(bf16_t));
  bf16_t* W2_1T  = (bf16_t*)alloc(128 * 128 * sizeof(bf16_t));
  bf16_t* Wg1T   = (bf16_t*)alloc(128 * 128 * sizeof(bf16_t));

  // Weight prep (bf16 transposed copies)
  k_conv_t<<<64, 256, 0, stream>>>(W1_0, W1_0T);
  k_conv_t<<<64, 256, 0, stream>>>(W2_0, W2_0T);
  k_conv_t<<<64, 256, 0, stream>>>(W1_1, W1_1T);
  k_conv_t<<<64, 256, 0, stream>>>(W2_1, W2_1T);
  k_conv_t<<<64, 256, 0, stream>>>(Wg1, Wg1T);

  const long long nfeat = (long long)N_NODES * 128;
  const int mlp_blocks  = 625;                 // 625 * 10 waves * 16 rows == N
  const int edge_blocks = (N_EDGES + 7) / 8;   // 200000

  // Init
  k_zero<<<2048, 256, 0, stream>>>(agg0, nfeat);
  k_zero<<<64, 256, 0, stream>>>(pooled, (long long)N_GRAPH * 128);
  k_init_pool<<<1, 128, 0, stream>>>(gmax, denom);

  // Layer 0
  k_edge<<<edge_blocks, 256, 0, stream>>>(x, ei, ea, We0, be0, agg0);
  k_gine_mlp<<<mlp_blocks, 320, 0, stream>>>(x, agg0, eps0, W1_0T, b1_0,
                                             W2_0T, b2_0, lg0, lb0, agg0);
  // Layer 1
  k_zero<<<2048, 256, 0, stream>>>(agg1, nfeat);
  k_edge<<<edge_blocks, 256, 0, stream>>>(agg0, ei, ea, We1, be1, agg1);
  k_gine_mlp<<<mlp_blocks, 320, 0, stream>>>(agg0, agg1, eps1, W1_1T, b1_1,
                                             W2_1T, b2_1, lg1, lb1, agg1);

  // Attentional pooling
  k_gate<<<mlp_blocks, 320, 0, stream>>>(agg1, Wg1T, bg1, Wg2, bg2, batch, gate, gmax);
  k_softmax1<<<(N_NODES + 255) / 256, 256, 0, stream>>>(gate, gmax, batch, exb, denom);
  k_pool<<<(int)(((long long)N_NODES * 32 + 255) / 256), 256, 0, stream>>>(
      agg1, exb, denom, batch, pooled);

  // Classifier
  k_classifier<<<N_GRAPH, 256, 0, stream>>>(pooled, gf, cg, cb, Wc1, bc1,
                                            Wc2, bc2, Wc3, bc3, (float*)d_out);
}
